// obj_semamtic_feat_42099269435831
// MI455X (gfx1250) — matrix operand order repaired and checked by reading, not compiled, not run on hardware
//
#include <hip/hip_runtime.h>

// ---------------------------------------------------------------------------
// MI455X (gfx1250). Big GEMMs via v_wmma_f32_16x16x32_bf16 (f32 accumulate).
// All f32->bf16 conversion is hoisted into streaming convert kernels so the
// GEMM inner loop is pure b128 loads + WMMA (previous version was VALU-bound
// on per-element bf16 conversion).
// ---------------------------------------------------------------------------

typedef __attribute__((ext_vector_type(16))) __bf16 v16bf;
typedef __attribute__((ext_vector_type(8)))  __bf16 v8bf;
typedef __attribute__((ext_vector_type(8)))  float  v8f;

// f32 -> bf16 round-to-nearest-even, bit-level (no __bf16 arithmetic needed)
__device__ inline __bf16 f32_to_bf16(float f) {
  unsigned u = __builtin_bit_cast(unsigned, f);
  u += 0x7FFFu + ((u >> 16) & 1u);
  unsigned short h = (unsigned short)(u >> 16);
  return __builtin_bit_cast(__bf16, h);
}
__device__ inline __bf16 bf16_zero() {
  return __builtin_bit_cast(__bf16, (unsigned short)0);
}
__device__ inline __bf16 to_bf16(float f)  { return f32_to_bf16(f); }
__device__ inline __bf16 to_bf16(__bf16 b) { return b; }

union V16 { v16bf v; v8bf h[2]; };

// A fragment: 16x32 bf16, ISA 7.12.2 layout. Row M = lane&15 (both halves);
// VGPR p: k = 2p + (p>=4?8:0) + half*8  =>  two contiguous 8-element chunks:
//   [k0 + half*8, +8) and [k0 + 16 + half*8, +8)   (16B-aligned)
__device__ inline v16bf load_a_frag(const __bf16* __restrict__ A, size_t arow,
                                    int k0, int half) {
  V16 u;
  u.h[0] = *(const v8bf*)(A + arow + k0 + half * 8);
  u.h[1] = *(const v8bf*)(A + arow + k0 + 16 + half * 8);
  return u.v;
}
__device__ inline v16bf load_a_tail(const __bf16* __restrict__ A, size_t arow,
                                    int k0, int half, int K) {
  v16bf a;
#pragma unroll
  for (int p = 0; p < 8; ++p) {
    int k = k0 + 2 * p + (p >= 4 ? 8 : 0) + half * 8;
    a[2 * p]     = (k     < K) ? A[arow + k]     : bf16_zero();
    a[2 * p + 1] = (k + 1 < K) ? A[arow + k + 1] : bf16_zero();
  }
  return a;
}

// B fragment: 32x16, B[k][n] = W[n][k] (W row-major NxK bf16). Col N = lane&15;
// VGPR p: k = 2p + (p>=4?8:0) + half*8  =>  two contiguous 8-element chunks:
//   [k0 + half*8, +8) and [k0 + 16 + half*8, +8)   (16B-aligned)
__device__ inline v16bf load_b_frag(const __bf16* __restrict__ W, size_t wrow,
                                    int k0, int half) {
  V16 u;
  u.h[0] = *(const v8bf*)(W + wrow + k0 + half * 8);
  u.h[1] = *(const v8bf*)(W + wrow + k0 + 16 + half * 8);
  return u.v;
}
__device__ inline v16bf load_b_tail(const __bf16* __restrict__ W, size_t wrow,
                                    int k0, int half, int K) {
  v16bf b;
#pragma unroll
  for (int p = 0; p < 8; ++p) {
    int k = k0 + 2 * p + (p >= 4 ? 8 : 0) + half * 8;
    b[2 * p]     = (k     < K) ? W[wrow + k]     : bf16_zero();
    b[2 * p + 1] = (k + 1 < K) ? W[wrow + k + 1] : bf16_zero();
  }
  return b;
}

// C = act(A[MxK](bf16) * W[NxK](bf16)^T + bias). One wave = 16 rows x NT*16 cols.
// Block = 128 threads = 4 waves = 4 consecutive M-tiles.
template <typename OT, bool RELU, int NT>
__global__ __launch_bounds__(128) void gemm_bf16_wmma(
    const __bf16* __restrict__ A, int lda,
    const __bf16* __restrict__ W, int ldw,
    const float* __restrict__ bias,
    OT* __restrict__ C, int ldc,
    int Mtiles, int N, int K) {
  const int lane  = threadIdx.x & 31;
  const int wave  = threadIdx.x >> 5;
  const int mtile = blockIdx.x * 4 + wave;
  if (mtile >= Mtiles) return;            // wave-uniform: EXEC all-ones at WMMAs
  const int half = lane >> 4;
  const int l15  = lane & 15;
  const size_t arow = (size_t)(mtile * 16 + l15) * (size_t)lda;
  const int ntile0 = blockIdx.y * NT;

  v8f acc[NT];
#pragma unroll
  for (int t = 0; t < NT; ++t)
#pragma unroll
    for (int r = 0; r < 8; ++r) acc[t][r] = 0.0f;

  int wrow[NT];
#pragma unroll
  for (int t = 0; t < NT; ++t) {
    int n = (ntile0 + t) * 16 + l15;
    if (n >= N) n = N - 1;                // clamp; stores masked in epilogue
    wrow[t] = n * ldw;
  }

  const int Kfull = K & ~31;
  for (int k0 = 0; k0 < Kfull; k0 += 32) {
    v16bf a = load_a_frag(A, arow, k0, half);
#pragma unroll
    for (int t = 0; t < NT; ++t) {
      v16bf b = load_b_frag(W, (size_t)wrow[t], k0, half);
      acc[t] = __builtin_amdgcn_wmma_f32_16x16x32_bf16(
          false, a, false, b, (short)0, acc[t], false, false);
    }
  }
  if (Kfull < K) {                        // ragged K tail (K = 200, 600)
    v16bf a = load_a_tail(A, arow, Kfull, half, K);
#pragma unroll
    for (int t = 0; t < NT; ++t) {
      v16bf b = load_b_tail(W, (size_t)wrow[t], Kfull, half, K);
      acc[t] = __builtin_amdgcn_wmma_f32_16x16x32_bf16(
          false, a, false, b, (short)0, acc[t], false, false);
    }
  }

  // Epilogue: C VGPR r holds (m = r + 8*half, n = lane&15)
#pragma unroll
  for (int t = 0; t < NT; ++t) {
    int n = (ntile0 + t) * 16 + l15;
    if (n >= N) continue;
    float bn = bias[n];
#pragma unroll
    for (int r = 0; r < 8; ++r) {
      int m = mtile * 16 + r + half * 8;
      float v = acc[t][r] + bn;
      if (RELU) v = v > 0.0f ? v : 0.0f;
      if constexpr (__is_same(OT, float)) {
        C[(size_t)m * (size_t)ldc + n] = v;
      } else {
        C[(size_t)m * (size_t)ldc + n] = f32_to_bf16(v);
      }
    }
  }
}

// Streaming f32 -> bf16 convert; 8 elements/thread; n must be a multiple of 8.
__global__ void convert_f32_bf16(const float* __restrict__ src,
                                 __bf16* __restrict__ dst, long long n8) {
  long long i = (long long)blockIdx.x * blockDim.x + threadIdx.x;
  if (i >= n8) return;
  long long base = i * 8;
  v8bf o;
#pragma unroll
  for (int j = 0; j < 8; ++j) o[j] = f32_to_bf16(src[base + j]);
  *(v8bf*)(dst + base) = o;
}

// Tiny exact-f32 linear for the embedding tables (51x200, 151x200; K=200)
__global__ void small_linear(const float* __restrict__ X, const float* __restrict__ W,
                             const float* __restrict__ B, float* __restrict__ Y,
                             int M, int N, int K) {
  int i = blockIdx.x * blockDim.x + threadIdx.x;
  if (i >= M * N) return;
  int m = i / N, n = i % N;
  float s = B[n];
  for (int k = 0; k < K; ++k) s += X[m * K + k] * W[n * K + k];
  Y[i] = s;
}

// out[r][0:200] = table[idx[r]][0:200]
__global__ void gather200(const float* __restrict__ table, const int* __restrict__ idx,
                          float* __restrict__ out, int rows) {
  int i = blockIdx.x * blockDim.x + threadIdx.x;
  if (i >= rows * 200) return;
  int r = i / 200, c = i % 200;
  out[(size_t)r * 200 + c] = table[(size_t)idx[r] * 200 + c];
}

// Fill side columns of a rows x 600 bf16 concat buffer from a rows x 200 table,
// indexed via rel_inds columns lcol / rcol (stride-3 int array).
template <typename ST>
__global__ void concat_sides(const ST* __restrict__ src, const int* __restrict__ rel_inds,
                             int lcol, int rcol, __bf16* __restrict__ dst, int rows) {
  int i = blockIdx.x * blockDim.x + threadIdx.x;
  if (i >= rows * 200) return;
  int r = i / 200, c = i % 200;
  int li = rel_inds[r * 3 + lcol];
  int ri = rel_inds[r * 3 + rcol];
  dst[(size_t)r * 600 + c]       = to_bf16(src[(size_t)li * 200 + c]);
  dst[(size_t)r * 600 + 400 + c] = to_bf16(src[(size_t)ri * 200 + c]);
}

static inline int cdiv(int a, int b) { return (a + b - 1) / b; }

extern "C" void kernel_launch(void* const* d_in, const int* in_sizes, int n_in,
                              void* d_out, int out_size, void* d_ws, size_t ws_size,
                              hipStream_t stream) {
  const float* obj_feats  = (const float*)d_in[1];
  const float* rel_feats  = (const float*)d_in[2];
  const int*   rel_inds   = (const int*)d_in[4];
  const int*   gt_rel_lab = (const int*)d_in[5];
  const int*   gt_obj_lab = (const int*)d_in[6];
  const float* obj_embed  = (const float*)d_in[7];
  const float* pred_embed = (const float*)d_in[8];
  const float *Wrs = (const float*)d_in[9],  *brs  = (const float*)d_in[10];
  const float *Wos = (const float*)d_in[11], *bos  = (const float*)d_in[12];
  const float *Wso1= (const float*)d_in[13], *bso1 = (const float*)d_in[14];
  const float *Wso2= (const float*)d_in[15], *bso2 = (const float*)d_in[16];
  const float *Wsoo= (const float*)d_in[17], *bsoo = (const float*)d_in[18];
  const float *Wp1 = (const float*)d_in[19], *bp1  = (const float*)d_in[20];
  const float *Wp2 = (const float*)d_in[21], *bp2  = (const float*)d_in[22];
  const float *Wpi = (const float*)d_in[23], *bpi  = (const float*)d_in[24];
  const float *Wpo1= (const float*)d_in[25], *bpo1 = (const float*)d_in[26];
  const float *Wpo2= (const float*)d_in[27], *bpo2 = (const float*)d_in[28];

  // d_out: tuple flattened in return order
  float* out          = (float*)d_out;
  float* s_o_out      = out;                 // 50000*200
  float* pred_out     = out + 10000000;      // 200000*200
  float* rel_semantic = out + 50000000;      // 200000*200
  float* obj_semantic = out + 90000000;      // 50000*200
  float* rel_sem_gt   = out + 100000000;     // 51*200
  float* obj_sem_gt   = out + 100010200;     // 151*200

  // -------- Workspace layout (lifetime-based reuse) --------
  char* ws = (char*)d_ws;
  size_t off = 0;
  auto take = [&](size_t bytes) -> char* {
    char* p = ws + off;
    off += (bytes + 127) & ~(size_t)127;
    return p;
  };
  // bf16 weights (live whole launch, ~3 MB -> L2-resident)
  __bf16* wso1_b = (__bf16*)take(512 * 1024 * 2);
  __bf16* wso2_b = (__bf16*)take(200 * 512 * 2);
  __bf16* wsoo_b = (__bf16*)take(200 * 200 * 2);
  __bf16* wp1_b  = (__bf16*)take(512 * 1024 * 2);
  __bf16* wp2_b  = (__bf16*)take(200 * 512 * 2);
  __bf16* wpi_b  = (__bf16*)take(200 * 600 * 2);
  __bf16* wpo1_b = (__bf16*)take(200 * 600 * 2);
  __bf16* wpo2_b = (__bf16*)take(200 * 200 * 2);
  // R1: objA (102.4 MB, dead after G1) then relA (409.6 MB, converted after G1)
  char* R1 = take(409600000);
  __bf16* objA = (__bf16*)R1;
  __bf16* relA = (__bf16*)R1;
  // R2: H1 (51.2 MB, G1->G2) then concat1 (240 MB, G5->G6)
  char* R2 = take(240000000);
  __bf16* H1      = (__bf16*)R2;
  __bf16* concat1 = (__bf16*)R2;
  // R3: H2 (204.8 MB, G4->G5) then concat2 (240 MB, G6->G7)
  char* R3 = take(240000000);
  __bf16* H2      = (__bf16*)R3;
  __bf16* concat2 = (__bf16*)R3;
  // R4: s_o_inter (20 MB, G2->sides1) then hidden (80 MB, G7->G8)
  char* R4 = take(80000000);
  __bf16* s_o_inter = (__bf16*)R4;
  __bf16* hidden    = (__bf16*)R4;

  const dim3 blk128(128), blk256(256);
  auto conv = [&](const float* s, __bf16* d, long long n) {
    convert_f32_bf16<<<cdiv((int)(n / 8), 256), blk256, 0, stream>>>(s, d, n / 8);
  };

  // Weight + obj_feats conversions (memory-bound; ~317 MB total traffic)
  conv(Wso1, wso1_b, 512 * 1024);
  conv(Wso2, wso2_b, 200 * 512);
  conv(Wsoo, wsoo_b, 200 * 200);
  conv(Wp1,  wp1_b,  512 * 1024);
  conv(Wp2,  wp2_b,  200 * 512);
  conv(Wpi,  wpi_b,  200 * 600);
  conv(Wpo1, wpo1_b, 200 * 600);
  conv(Wpo2, wpo2_b, 200 * 200);
  conv(obj_feats, objA, 50000LL * 1024);

  // Tiny embedding linears (exact f32) + semantic gathers
  small_linear<<<cdiv(51 * 200, 256), blk256, 0, stream>>>(pred_embed, Wrs, brs, rel_sem_gt, 51, 200, 200);
  small_linear<<<cdiv(151 * 200, 256), blk256, 0, stream>>>(obj_embed, Wos, bos, obj_sem_gt, 151, 200, 200);
  gather200<<<cdiv(200000 * 200, 256), blk256, 0, stream>>>(rel_sem_gt, gt_rel_lab, rel_semantic, 200000);
  gather200<<<cdiv(50000 * 200, 256), blk256, 0, stream>>>(obj_sem_gt, gt_obj_lab, obj_semantic, 50000);

  // Object path: 50000 rows -> 3125 M-tiles -> 782 blocks of 4 waves
  gemm_bf16_wmma<__bf16, true, 8><<<dim3(782, 4), blk128, 0, stream>>>(
      objA, 1024, wso1_b, 1024, bso1, H1, 512, 3125, 512, 1024);          // G1
  gemm_bf16_wmma<__bf16, true, 8><<<dim3(782, 2), blk128, 0, stream>>>(
      H1, 512, wso2_b, 512, bso2, s_o_inter, 200, 3125, 200, 512);        // G2
  gemm_bf16_wmma<float, false, 8><<<dim3(782, 2), blk128, 0, stream>>>(
      s_o_inter, 200, wsoo_b, 200, bsoo, s_o_out, 200, 3125, 200, 200);   // G3

  // rel_feats conversion after G1 so relA can reuse objA's region
  conv(rel_feats, relA, 200000LL * 1024);

  // Relation path: 200000 rows -> 12500 M-tiles -> 3125 blocks
  gemm_bf16_wmma<__bf16, true, 8><<<dim3(3125, 4), blk128, 0, stream>>>(
      relA, 1024, wp1_b, 1024, bp1, H2, 512, 12500, 512, 1024);           // G4
  gemm_bf16_wmma<__bf16, true, 8><<<dim3(3125, 2), blk128, 0, stream>>>(
      H2, 512, wp2_b, 512, bp2, concat1 + 200, 600, 12500, 200, 512);     // G5
  concat_sides<__bf16><<<cdiv(200000 * 200, 256), blk256, 0, stream>>>(
      s_o_inter, rel_inds, 1, 1, concat1, 200000);  // ref uses sub_ind twice
  gemm_bf16_wmma<__bf16, true, 8><<<dim3(3125, 2), blk128, 0, stream>>>(
      concat1, 600, wpi_b, 600, bpi, concat2 + 200, 600, 12500, 200, 600); // G6
  concat_sides<float><<<cdiv(200000 * 200, 256), blk256, 0, stream>>>(
      s_o_out, rel_inds, 1, 2, concat2, 200000);
  gemm_bf16_wmma<__bf16, true, 8><<<dim3(3125, 2), blk128, 0, stream>>>(
      concat2, 600, wpo1_b, 600, bpo1, hidden, 200, 12500, 200, 600);     // G7
  gemm_bf16_wmma<float, false, 8><<<dim3(3125, 2), blk128, 0, stream>>>(
      hidden, 200, wpo2_b, 200, bpo2, pred_out, 200, 12500, 200, 200);    // G8

  (void)in_sizes; (void)n_in; (void)out_size; (void)ws_size;
}
